// FeedForwardQuantum_65481071395981
// MI455X (gfx1250) — matrix operand order repaired and checked by reading, not compiled, not run on hardware
//
#include <hip/hip_runtime.h>

// ---------------------------------------------------------------------------
// FeedForwardQuantum for MI455X (gfx1250, wave32, WMMA)
//   q   = cos(x[..., :8]) * cos(theta)          [M=32768, 8]   f32   (ws)
//   h   = relu(q @ W1^T)                        [M, F=4096]    bf16  (ws)
//   out = h @ W2^T                              [M, E=1024]    f32   (d_out)
// GEMM uses v_wmma_f32_16x16x32_bf16 with f32 accumulate.
// All memory ops forced to address_space(1) -> global_load/global_store.
// Fragment tile offsets are compile-time constants folded into the 24-bit
// instruction ioffset; only 2 pointer increments per K-iteration.
// ---------------------------------------------------------------------------

#define GLOBAL __attribute__((address_space(1)))

typedef __attribute__((ext_vector_type(16))) __bf16 bf16x16;
typedef __attribute__((ext_vector_type(8)))  __bf16 bf16x8;
typedef __attribute__((ext_vector_type(8)))  float  floatx8;
typedef __attribute__((ext_vector_type(8)))  unsigned short u16x8;
typedef __attribute__((ext_vector_type(4)))  float  f32x4;

typedef GLOBAL const bf16x8* gbf8cp;
typedef GLOBAL const f32x4*  gf4cp;

#define M_ROWS 32768   // B*S
#define E_DIM  1024
#define F_DIM  4096
#define NQ     8

__device__ __forceinline__ unsigned short f32_to_bf16(float f) {
  unsigned int u = __builtin_bit_cast(unsigned int, f);
  u += 0x7FFFu + ((u >> 16) & 1u);   // round-to-nearest-even
  return (unsigned short)(u >> 16);
}

__device__ __forceinline__ float fast_cos(float x) {
  // v_cos_f32 takes revolutions: cos(2*pi*t); scale by 1/(2*pi)
  return __builtin_amdgcn_cosf(x * 0.15915494309189535f);
}

// ------------------------- K0: q = cos(x)*cos(theta) ------------------------
__global__ void qk(const float* __restrict__ x, const float* __restrict__ theta,
                   float* __restrict__ q) {
  int m = blockIdx.x * blockDim.x + threadIdx.x;
  if (m >= M_ROWS) return;
  GLOBAL const float* xr = (GLOBAL const float*)(x + (size_t)m * E_DIM);
  GLOBAL const float* th = (GLOBAL const float*)theta;
  GLOBAL float*       qo = (GLOBAL float*)(q + (size_t)m * NQ);
#pragma unroll
  for (int n = 0; n < NQ; ++n)
    qo[n] = fast_cos(xr[n]) * fast_cos(th[n]);
}

// ------------------------- K1: W2 f32 -> bf16 -------------------------------
__global__ void w2k(const float* __restrict__ w2, unsigned short* __restrict__ w2b) {
  size_t base = ((size_t)blockIdx.x * blockDim.x + threadIdx.x) * 8;
  gf4cp p = (gf4cp)(w2 + base);
  f32x4 a = p[0], b = p[1];
  u16x8 o;
  o[0] = f32_to_bf16(a[0]); o[1] = f32_to_bf16(a[1]);
  o[2] = f32_to_bf16(a[2]); o[3] = f32_to_bf16(a[3]);
  o[4] = f32_to_bf16(b[0]); o[5] = f32_to_bf16(b[1]);
  o[6] = f32_to_bf16(b[2]); o[7] = f32_to_bf16(b[3]);
  *(GLOBAL u16x8*)(w2b + base) = o;
}

// ------------------------- K2: h = relu(q @ W1^T), bf16 ---------------------
__global__ void hk(const float* __restrict__ q, const float* __restrict__ W1,
                   unsigned short* __restrict__ h) {
  int gid = blockIdx.x * blockDim.x + threadIdx.x;  // M_ROWS * (F/8) threads
  int m  = gid >> 9;            // / (F_DIM/8 = 512)
  int f0 = (gid & 511) * 8;
  gf4cp qr = (gf4cp)(q + (size_t)m * NQ);
  f32x4 q0 = qr[0], q1 = qr[1];
  u16x8 o;
#pragma unroll
  for (int j = 0; j < 8; ++j) {
    gf4cp w = (gf4cp)(W1 + (size_t)(f0 + j) * NQ);
    f32x4 w0 = w[0], w1 = w[1];
    float s = q0[0] * w0[0] + q0[1] * w0[1] + q0[2] * w0[2] + q0[3] * w0[3] +
              q1[0] * w1[0] + q1[1] * w1[1] + q1[2] * w1[2] + q1[3] * w1[3];
    s = fmaxf(s, 0.0f);
    o[j] = f32_to_bf16(s);
  }
  *(GLOBAL u16x8*)(h + (size_t)m * F_DIM + f0) = o;
}

// ------------------------- K3: out = h @ W2b^T  (WMMA bf16) -----------------
// Block tile 128x256, 8 waves (2x4), wave tile 64x64 = 4x4 x (16x16x32 WMMA).
// A (h) is [M,K] row-major; B (w2b) is [N,K] row-major so b[k,n]=w2b[n][k]
// is contiguous along K per lane -- matches both ISA fragment layouts:
//   A 16x32: lane l16 = M row; K = lh*8 + [0,8) and +16
//   B 32x16: lane l16 = N col; K = lh*8 + [0,8) and +16
__global__ void __launch_bounds__(256)
gemmk(const __bf16* __restrict__ A, const __bf16* __restrict__ Bm,
      float* __restrict__ C) {
  const int K = F_DIM;   // 4096
  const int N = E_DIM;   // 1024
  const int TILE_B = 16 * F_DIM;  // elements between adjacent 16-row tiles

  int lane = threadIdx.x & 31;
  int wave = threadIdx.x >> 5;
  int l16  = lane & 15;
  int lh   = lane >> 4;          // 0: lanes 0-15, 1: lanes 16-31
  int wrow = wave >> 2;          // 0..1
  int wcol = wave & 3;           // 0..3
  int m0 = blockIdx.y * 128 + wrow * 64;
  int n0 = blockIdx.x * 256 + wcol * 64;

  floatx8 acc[4][4];
#pragma unroll
  for (int i = 0; i < 4; ++i)
#pragma unroll
    for (int j = 0; j < 4; ++j) {
      floatx8 z = {0.f, 0.f, 0.f, 0.f, 0.f, 0.f, 0.f, 0.f};
      acc[i][j] = z;
    }

  // single per-lane base pointer per matrix; mt/nt tiles reached via
  // constant offsets (mt*16*K*2 bytes = 131072*mt, fits 24-bit ioffset)
  GLOBAL const __bf16* aB =
      (GLOBAL const __bf16*)(A + (size_t)(m0 + l16) * K + lh * 8);
  GLOBAL const __bf16* bB =
      (GLOBAL const __bf16*)(Bm + (size_t)(n0 + l16) * K + lh * 8);

  for (int kk = 0; kk < K; kk += 32) {
    bf16x16 af[4], bfr[4];
#pragma unroll
    for (int mt = 0; mt < 4; ++mt) {
      union { bf16x16 v; bf16x8 h[2]; } u;
      gbf8cp p = (gbf8cp)(aB + mt * TILE_B);
      u.h[0] = p[0];           // K = kk + lh*8 + [0..8)
      u.h[1] = p[2];           // K = kk + lh*8 + 16 + [0..8)
      af[mt] = u.v;
      // prefetch the streaming (HBM-resident) A operand ~4 iters ahead
      __builtin_prefetch((const void*)(aB + mt * TILE_B + 128), 0, 1);
    }
#pragma unroll
    for (int nt = 0; nt < 4; ++nt) {
      union { bf16x16 v; bf16x8 h[2]; } u;
      gbf8cp p = (gbf8cp)(bB + nt * TILE_B);
      u.h[0] = p[0];           // K = kk + lh*8 + [0..8)
      u.h[1] = p[2];           // K = kk + lh*8 + 16 + [0..8)
      bfr[nt] = u.v;
    }
#pragma unroll
    for (int mt = 0; mt < 4; ++mt)
#pragma unroll
      for (int nt = 0; nt < 4; ++nt)
        acc[mt][nt] = __builtin_amdgcn_wmma_f32_16x16x32_bf16(
            false, af[mt], false, bfr[nt], (short)0, acc[mt][nt], false, false);
    aB += 32;
    bB += 32;
  }

  // D layout: VGPR r -> M = mtile + lh*8 + r, N = ntile + l16
#pragma unroll
  for (int mt = 0; mt < 4; ++mt)
#pragma unroll
    for (int nt = 0; nt < 4; ++nt) {
      int n = n0 + nt * 16 + l16;
#pragma unroll
      for (int r = 0; r < 8; ++r) {
        int m = m0 + mt * 16 + lh * 8 + r;
        *(GLOBAL float*)(C + (size_t)m * N + n) = acc[mt][nt][r];
      }
    }
}

// ---------------------------------------------------------------------------
extern "C" void kernel_launch(void* const* d_in, const int* in_sizes, int n_in,
                              void* d_out, int out_size, void* d_ws, size_t ws_size,
                              hipStream_t stream) {
  const float* x     = (const float*)d_in[0];
  const float* theta = (const float*)d_in[1];
  const float* W1    = (const float*)d_in[2];
  const float* W2    = (const float*)d_in[3];
  float* out = (float*)d_out;

  char* ws = (char*)d_ws;
  float*          qbuf = (float*)ws;                                 // 1 MB
  unsigned short* w2b  = (unsigned short*)(ws + ((size_t)1  << 20)); // 8 MB
  unsigned short* hbuf = (unsigned short*)(ws + ((size_t)16 << 20)); // 256 MB

  qk <<<(M_ROWS + 255) / 256, 256, 0, stream>>>(x, theta, qbuf);
  w2k<<<(E_DIM * F_DIM / 8) / 256, 256, 0, stream>>>(W2, w2b);
  hk <<<(M_ROWS * (F_DIM / 8)) / 256, 256, 0, stream>>>(qbuf, W1, hbuf);
  gemmk<<<dim3(E_DIM / 256, M_ROWS / 128), 256, 0, stream>>>(
      (const __bf16*)hbuf, (const __bf16*)w2b, out);
}
